// TopoAttention_3229815407291
// MI455X (gfx1250) — matrix operand order repaired and checked by reading, not compiled, not run on hardware
//
#include <hip/hip_runtime.h>
#include <hip/hip_bf16.h>

// ---------------------------------------------------------------------------
// TopoAttention for MI455X (gfx1250, wave32, WMMA + TDM).
//   * All dense math through v_wmma_f32_16x16x32_f16 (f16 operands, f32 acc).
//   * Tile staging via TENSOR_LOAD_TO_LDS (TDM DMA) with double buffering;
//     wave 0 issues descriptors, s_wait_tensorcnt + workgroup barrier syncs.
//   * Flash-style fused attention, renormalized masked softmax:
//       sparse = exp(s-m)*mask / (MS + 1e-8*Z),  Z/MS tracked online.
//   * Edge mask as N x N/32 bitmask; col_mask via exact O(N^2) rank top-k.
//   * mask_exp (537 MB) written straight from bitmask+colmask, float4 stores.
// ---------------------------------------------------------------------------

typedef __attribute__((ext_vector_type(16))) _Float16 v16h;
typedef __attribute__((ext_vector_type(8)))  float    v8f;
typedef __attribute__((ext_vector_type(4)))  unsigned tdm_u4;
typedef __attribute__((ext_vector_type(8)))  int      tdm_i8;
typedef __attribute__((ext_vector_type(4)))  int      tdm_i4;

#define B_    4
#define N_    2048
#define D_    512
#define H_    8
#define HD_   64
#define DH_   256
#define E_    65536
#define M_    (B_ * N_)
#define KTOP_ 1024
#define NW32  (N_ / 32)

#ifndef __has_builtin
#define __has_builtin(x) 0
#endif
#if __has_builtin(__builtin_amdgcn_tensor_load_to_lds) && \
    __has_builtin(__builtin_amdgcn_s_wait_tensorcnt)
#define HAVE_TDM 1
#else
#define HAVE_TDM 0
#endif

#define WMMA_F16(a, b, c) \
  __builtin_amdgcn_wmma_f32_16x16x32_f16(false, (a), false, (b), (short)0, (c), false, false)

// --- TDM helpers -------------------------------------------------------------
// Low 32 bits of a flat pointer to LDS are the LDS byte address (ISA 10.2:
// LDS aperture).
__device__ __forceinline__ unsigned lds_addr_of(const void* p) {
  return (unsigned)(unsigned long long)p;
}

#if HAVE_TDM
// 2D tile load: tile_w elements per row (data_size=2B), tile_h rows,
// row stride = stride_e elements. Written contiguously to LDS at `lds`.
__device__ __forceinline__ void tdm_load_2d(unsigned lds, const void* gp,
                                            int tile_w, int tile_h, int stride_e) {
  unsigned long long ga = (unsigned long long)gp;
  tdm_u4 g0;
  g0[0] = 1u;                                   // count=1, user descriptor
  g0[1] = lds;                                  // lds_addr (bytes)
  g0[2] = (unsigned)(ga & 0xffffffffu);         // global_addr[31:0]
  g0[3] = (unsigned)(ga >> 32) | (2u << 30);    // global_addr[56:32] | type=2
  tdm_i8 g1;
  g1[0] = 0x00010000;                           // data_size=1 (2 bytes), no mcast
  g1[1] = (tile_w & 0xffff) << 16;              // tensor_dim0[15:0]
  g1[2] = (tile_h & 0xffff) << 16;              // tensor_dim1[15:0] (dim0 hi = 0)
  g1[3] = (tile_w & 0xffff) << 16;              // tile_dim0 (dim1 hi = 0)
  g1[4] = (tile_h & 0xffff);                    // tile_dim1, tile_dim2 = 0
  g1[5] = stride_e;                             // tensor_dim0_stride[31:0]
  g1[6] = 0;
  g1[7] = 0;
  tdm_i4 z4 = {0, 0, 0, 0};
#if __clang_major__ >= 23
  tdm_i8 z8 = {0, 0, 0, 0, 0, 0, 0, 0};
  __builtin_amdgcn_tensor_load_to_lds(g0, g1, z4, z4, z8, 0);
#else
  __builtin_amdgcn_tensor_load_to_lds(g0, g1, z4, z4, 0);
#endif
}
#endif

// --- fragment loaders matching the documented gfx1250 VGPR layouts -----------
__device__ __forceinline__ v16h load_A16(const _Float16* a, int lda, int m0, int k0, int lane) {
  int m  = m0 + (lane & 15);
  int hs = (lane >> 4) & 1;
  v16h r;
#pragma unroll
  for (int e = 0; e < 16; ++e) {
    int k = k0 + ((e >> 3) << 4) + hs * 8 + ((e >> 1) & 3) * 2 + (e & 1);
    r[e] = a[m * lda + k];
  }
  return r;
}

__device__ __forceinline__ v16h load_B16(const _Float16* b, int sk, int sn, int k0, int n0, int lane) {
  int n  = n0 + (lane & 15);
  int hs = (lane >> 4) & 1;
  v16h r;
#pragma unroll
  for (int e = 0; e < 16; ++e) {
    int k = k0 + ((e >> 3) << 4) + hs * 8 + (e & 7);
    r[e] = b[k * sk + n * sn];
  }
  return r;
}

// --- small utility kernels ---------------------------------------------------
__global__ void cvt_f32_f16(const float* __restrict__ in, _Float16* __restrict__ out, int n) {
  int i = blockIdx.x * blockDim.x + threadIdx.x;
  if (i < n) out[i] = (_Float16)in[i];
}

__global__ void zero_u32(unsigned* __restrict__ p, int n) {
  int i = blockIdx.x * blockDim.x + threadIdx.x;
  if (i < n) p[i] = 0u;
}

__global__ void scatter_edges(const int* __restrict__ ei, unsigned* __restrict__ ebm) {
  int e = blockIdx.x * blockDim.x + threadIdx.x;
  if (e < E_) {
    int r = ei[e];
    int c = ei[E_ + e];
    atomicOr(&ebm[r * NW32 + (c >> 5)], 1u << (c & 31));
  }
}

// --- WMMA GEMM: C[M,Nd] = A[M,Kd] @ W[Kd,Nd] + bias --------------------------
// 128 threads (4 waves), 64x64 tile, double-buffered TDM tile staging.
template <int RELU, int O16>
__global__ __launch_bounds__(128) void gemm_wmma(
    const _Float16* __restrict__ A, const _Float16* __restrict__ W,
    const float* __restrict__ bias, _Float16* __restrict__ o16,
    float* __restrict__ o32, int Kd, int Nd) {
  __shared__ _Float16 At[2][64 * 32];
  __shared__ _Float16 Wt[2][32 * 64];
  int tid = threadIdx.x, lane = tid & 31, wv = tid >> 5;
  int wm = wv >> 1, wn = wv & 1;
  int rowbase = blockIdx.x * 64, colbase = blockIdx.y * 64;
  int nk = Kd >> 5;

  v8f acc[4] = {{}, {}, {}, {}};

#if HAVE_TDM
  auto issue = [&](int kt, int bi) {
    tdm_load_2d(lds_addr_of(&At[bi][0]), &A[(size_t)rowbase * Kd + kt * 32], 32, 64, Kd);
    tdm_load_2d(lds_addr_of(&Wt[bi][0]), &W[(size_t)(kt * 32) * Nd + colbase], 64, 32, Nd);
  };
  if (wv == 0) issue(0, 0);
#endif

  for (int kt = 0; kt < nk; ++kt) {
    int cur = kt & 1;
#if HAVE_TDM
    if (wv == 0) {
      if (kt + 1 < nk) {
        issue(kt + 1, cur ^ 1);
        __builtin_amdgcn_s_wait_tensorcnt((short)2);  // current pair landed
      } else {
        __builtin_amdgcn_s_wait_tensorcnt((short)0);
      }
    }
#else
#pragma unroll
    for (int t = 0; t < 16; ++t) {
      int idx = tid * 16 + t;
      At[cur][idx] = A[(size_t)(rowbase + (idx >> 5)) * Kd + kt * 32 + (idx & 31)];
      Wt[cur][idx] = W[(size_t)(kt * 32 + (idx >> 6)) * Nd + colbase + (idx & 63)];
    }
#endif
    __syncthreads();
    const _Float16* Ac = &At[cur][0];
    const _Float16* Wc = &Wt[cur][0];
    v16h a0 = load_A16(Ac, 32, wm * 32, 0, lane);
    v16h a1 = load_A16(Ac, 32, wm * 32 + 16, 0, lane);
    v16h b0 = load_B16(Wc, 64, 1, 0, wn * 32, lane);
    v16h b1 = load_B16(Wc, 64, 1, 0, wn * 32 + 16, lane);
    acc[0] = WMMA_F16(a0, b0, acc[0]);
    acc[1] = WMMA_F16(a0, b1, acc[1]);
    acc[2] = WMMA_F16(a1, b0, acc[2]);
    acc[3] = WMMA_F16(a1, b1, acc[3]);
    __syncthreads();
  }

  int hs = lane >> 4, nl = lane & 15;
#pragma unroll
  for (int ms = 0; ms < 2; ++ms)
#pragma unroll
    for (int ns = 0; ns < 2; ++ns) {
      v8f a = acc[ms * 2 + ns];
#pragma unroll
      for (int c = 0; c < 8; ++c) {
        int row = rowbase + wm * 32 + ms * 16 + c + 8 * hs;
        int col = colbase + wn * 32 + ns * 16 + nl;
        float v = a[c] + bias[col];
        if (RELU) v = v > 0.f ? v : 0.f;
        if (O16) o16[(size_t)row * Nd + col] = (_Float16)v;
        else     o32[(size_t)row * Nd + col] = v;
      }
    }
}

// --- topo score tail ---------------------------------------------------------
__global__ void topo_reduce(const _Float16* __restrict__ hid, const float* __restrict__ Wt2,
                            const float* __restrict__ bt2, float* __restrict__ scores) {
  int n = blockIdx.x * blockDim.x + threadIdx.x;
  if (n >= M_) return;
  float s = bt2[0];
  const _Float16* h = hid + (size_t)n * DH_;
  for (int j = 0; j < DH_; ++j) s += (float)h[j] * Wt2[j];
  scores[n] = s;
}

// --- exact top-k via rank (ties: lower index first, matching lax.top_k) ------
__global__ __launch_bounds__(256) void rank_topk(const float* __restrict__ scores,
                                                 float* __restrict__ colmask) {
  __shared__ float sc[N_];
  int b = blockIdx.x, t = threadIdx.x;
  for (int j = t; j < N_; j += 256) sc[j] = scores[b * N_ + j];
  __syncthreads();
  for (int i = t; i < N_; i += 256) {
    float si = sc[i];
    int rank = 0;
    for (int j = 0; j < N_; ++j) {
      float sj = sc[j];
      rank += (sj > si) || (sj == si && j < i);
    }
    colmask[b * N_ + i] = (rank < KTOP_) ? 1.f : 0.f;
  }
}

// --- write mask_exp [B,H,N,N] ------------------------------------------------
__global__ __launch_bounds__(256) void write_mask(const unsigned* __restrict__ ebm,
                                                  const float* __restrict__ colmask,
                                                  float* __restrict__ maskout) {
  __shared__ float mrow[N_];
  int i = blockIdx.x, b = blockIdx.y, t = threadIdx.x;
  const unsigned* er = ebm + (size_t)i * NW32;
  for (int j = t; j < N_; j += 256) {
    int eb = (er[j >> 5] >> (j & 31)) & 1;
    mrow[j] = (eb || colmask[b * N_ + j] > 0.f) ? 1.f : 0.f;
  }
  __syncthreads();
  const float4* src = (const float4*)mrow;
#pragma unroll
  for (int h = 0; h < H_; ++h) {
    float4* d4 = (float4*)(maskout + ((size_t)(b * H_ + h) * N_ + i) * N_);
    for (int j = t; j < N_ / 4; j += 256) d4[j] = src[j];
  }
}

// --- fused flash attention with masked renormalization -----------------------
__global__ __launch_bounds__(128) void attn_wmma(
    const _Float16* __restrict__ Qh, const _Float16* __restrict__ Kh,
    const _Float16* __restrict__ Vh, const unsigned* __restrict__ ebm,
    const float* __restrict__ colmask, _Float16* __restrict__ Oh) {
  __shared__ _Float16 Qt[64 * 64];
  __shared__ _Float16 Kt[2][64 * 64];
  __shared__ _Float16 Vt[2][64 * 64];
  __shared__ _Float16 Pt[4][16 * 64];
  __shared__ float    cmt[64];
  __shared__ unsigned et[64][2];

  int tid = threadIdx.x, lane = tid & 31, wv = tid >> 5;
  int m0 = blockIdx.x * 64, h = blockIdx.y, b = blockIdx.z;
  int hs = lane >> 4, nl = lane & 15;
  const int nj = N_ / 64;

#if HAVE_TDM
  auto issueKV = [&](int jt, int bi) {
    size_t base = (size_t)(b * N_ + jt * 64) * D_ + h * HD_;
    tdm_load_2d(lds_addr_of(&Kt[bi][0]), &Kh[base], 64, 64, D_);
    tdm_load_2d(lds_addr_of(&Vt[bi][0]), &Vh[base], 64, 64, D_);
  };
  if (wv == 0) {
    tdm_load_2d(lds_addr_of(&Qt[0]), &Qh[(size_t)(b * N_ + m0) * D_ + h * HD_], 64, 64, D_);
    issueKV(0, 0);
  }
#else
  for (int idx = tid; idx < 64 * 64; idx += 128) {
    int r = idx >> 6, d = idx & 63;
    Qt[idx] = Qh[(size_t)(b * N_ + m0 + r) * D_ + h * HD_ + d];
  }
#endif

  float mx[8], Z[8], MS[8];
  v8f oacc[4] = {{}, {}, {}, {}};
#pragma unroll
  for (int c = 0; c < 8; ++c) { mx[c] = -1e30f; Z[c] = 0.f; MS[c] = 0.f; }
  v16h qf0, qf1;

  for (int jt = 0; jt < nj; ++jt) {
    int cur = jt & 1;
#if HAVE_TDM
    if (wv == 0) {
      if (jt + 1 < nj) {
        issueKV(jt + 1, cur ^ 1);
        __builtin_amdgcn_s_wait_tensorcnt((short)2);  // Q + current K/V landed
      } else {
        __builtin_amdgcn_s_wait_tensorcnt((short)0);
      }
    }
#else
    for (int idx = tid; idx < 64 * 64; idx += 128) {
      int r = idx >> 6, d = idx & 63;
      size_t g = (size_t)(b * N_ + jt * 64 + r) * D_ + h * HD_ + d;
      Kt[cur][idx] = Kh[g];
      Vt[cur][idx] = Vh[g];
    }
#endif
    if (tid < 64) cmt[tid] = colmask[b * N_ + jt * 64 + tid];
    {
      int r = tid >> 1, wd = tid & 1;
      et[r][wd] = ebm[(size_t)(m0 + r) * NW32 + ((jt * 64) >> 5) + wd];
    }
    if (jt + 1 < nj)  // gfx1250 global_prefetch_b8 for the small side streams
      __builtin_prefetch(&colmask[b * N_ + (jt + 1) * 64 + (tid & 63)], 0, 1);
    __syncthreads();
    if (jt == 0) {
      qf0 = load_A16(Qt, 64, wv * 16, 0, lane);
      qf1 = load_A16(Qt, 64, wv * 16, 32, lane);
    }
    const _Float16* Kc = &Kt[cur][0];
    const _Float16* Vc = &Vt[cur][0];

    // S = (Q . K^T) * 1/sqrt(64); K read transposed out of LDS (sk=1, sn=64)
    v8f s[4];
#pragma unroll
    for (int ns = 0; ns < 4; ++ns) {
      v16h bk0 = load_B16(Kc, 1, 64, 0, ns * 16, lane);
      v16h bk1 = load_B16(Kc, 1, 64, 32, ns * 16, lane);
      v8f a = {};
      a = WMMA_F16(qf0, bk0, a);
      a = WMMA_F16(qf1, bk1, a);
#pragma unroll
      for (int c = 0; c < 8; ++c) a[c] *= 0.125f;
      s[ns] = a;
    }

    // tile row max -> cross-lane (width 16 keeps the two row-halves separate)
    float tmax[8];
#pragma unroll
    for (int c = 0; c < 8; ++c)
      tmax[c] = fmaxf(fmaxf(s[0][c], s[1][c]), fmaxf(s[2][c], s[3][c]));
#pragma unroll
    for (int off = 1; off < 16; off <<= 1)
#pragma unroll
      for (int c = 0; c < 8; ++c) tmax[c] = fmaxf(tmax[c], __shfl_xor(tmax[c], off, 16));

    float zadd[8], msadd[8];
#pragma unroll
    for (int c = 0; c < 8; ++c) {
      float nm = fmaxf(mx[c], tmax[c]);
      float rs = __expf(mx[c] - nm);
      Z[c] *= rs; MS[c] *= rs;
#pragma unroll
      for (int ds = 0; ds < 4; ++ds) oacc[ds][c] *= rs;
      mx[c] = nm;
      zadd[c] = 0.f; msadd[c] = 0.f;
    }

#pragma unroll
    for (int ns = 0; ns < 4; ++ns) {
      int jl = ns * 16 + nl;
      float cmv = cmt[jl];
#pragma unroll
      for (int c = 0; c < 8; ++c) {
        int il = wv * 16 + c + 8 * hs;
        float p = __expf(s[ns][c] - mx[c]);
        zadd[c] += p;
        unsigned eb = (et[il][jl >> 5] >> (jl & 31)) & 1u;
        float mk = (eb || cmv > 0.f) ? 1.f : 0.f;
        float pm = p * mk;
        msadd[c] += pm;
        Pt[wv][(c + 8 * hs) * 64 + jl] = (_Float16)pm;
      }
    }
#pragma unroll
    for (int off = 1; off < 16; off <<= 1)
#pragma unroll
      for (int c = 0; c < 8; ++c) {
        zadd[c] += __shfl_xor(zadd[c], off, 16);
        msadd[c] += __shfl_xor(msadd[c], off, 16);
      }
#pragma unroll
    for (int c = 0; c < 8; ++c) { Z[c] += zadd[c]; MS[c] += msadd[c]; }

    // O += P @ V (P via per-wave LDS to get the A-matrix lane layout)
    v16h p0 = load_A16(Pt[wv], 64, 0, 0, lane);
    v16h p1 = load_A16(Pt[wv], 64, 0, 32, lane);
#pragma unroll
    for (int ds = 0; ds < 4; ++ds) {
      v16h bv0 = load_B16(Vc, 64, 1, 0, ds * 16, lane);
      v16h bv1 = load_B16(Vc, 64, 1, 32, ds * 16, lane);
      oacc[ds] = WMMA_F16(p0, bv0, oacc[ds]);
      oacc[ds] = WMMA_F16(p1, bv1, oacc[ds]);
    }
    __syncthreads();
  }

#pragma unroll
  for (int c = 0; c < 8; ++c) {
    float rs = 1.f / (MS[c] + 1e-8f * Z[c]);
#pragma unroll
    for (int ds = 0; ds < 4; ++ds) oacc[ds][c] *= rs;
  }
#pragma unroll
  for (int ds = 0; ds < 4; ++ds)
#pragma unroll
    for (int c = 0; c < 8; ++c) {
      int row = m0 + wv * 16 + c + 8 * hs;
      int col = h * HD_ + ds * 16 + nl;
      Oh[(size_t)(b * N_ + row) * D_ + col] = (_Float16)oacc[ds][c];
    }
}

// ---------------------------------------------------------------------------
extern "C" void kernel_launch(void* const* d_in, const int* in_sizes, int n_in,
                              void* d_out, int out_size, void* d_ws, size_t ws_size,
                              hipStream_t stream) {
  const float* x   = (const float*)d_in[0];
  const int*   ei  = (const int*)d_in[1];
  const float* Wq  = (const float*)d_in[2];  const float* bq  = (const float*)d_in[3];
  const float* Wk  = (const float*)d_in[4];  const float* bk  = (const float*)d_in[5];
  const float* Wv  = (const float*)d_in[6];  const float* bv  = (const float*)d_in[7];
  const float* Wo  = (const float*)d_in[8];  const float* bo  = (const float*)d_in[9];
  const float* Wt1 = (const float*)d_in[10]; const float* bt1 = (const float*)d_in[11];
  const float* Wt2 = (const float*)d_in[12]; const float* bt2 = (const float*)d_in[13];

  char* ws = (char*)d_ws;
  size_t off = 0;
  auto take = [&](size_t bytes) -> char* {
    char* p = ws + off;
    off = (off + bytes + 255) & ~(size_t)255;
    return p;
  };
  _Float16* Xh   = (_Float16*)take((size_t)M_ * D_ * 2);
  _Float16* Wqh  = (_Float16*)take((size_t)D_ * D_ * 2);
  _Float16* Wkh  = (_Float16*)take((size_t)D_ * D_ * 2);
  _Float16* Wvh  = (_Float16*)take((size_t)D_ * D_ * 2);
  _Float16* Woh  = (_Float16*)take((size_t)D_ * D_ * 2);
  _Float16* Wt1h = (_Float16*)take((size_t)D_ * DH_ * 2);
  _Float16* Qh   = (_Float16*)take((size_t)M_ * D_ * 2);
  _Float16* Kh   = (_Float16*)take((size_t)M_ * D_ * 2);
  _Float16* Vh   = (_Float16*)take((size_t)M_ * D_ * 2);
  _Float16* Hid  = (_Float16*)take((size_t)M_ * DH_ * 2);
  _Float16* Oh   = (_Float16*)take((size_t)M_ * D_ * 2);
  float*    scr  = (float*)take((size_t)M_ * 4);
  float*    cms  = (float*)take((size_t)M_ * 4);
  unsigned* ebm  = (unsigned*)take((size_t)N_ * NW32 * 4);

  // 1) precision conversion
  cvt_f32_f16<<<(M_ * D_ + 255) / 256, 256, 0, stream>>>(x, Xh, M_ * D_);
  cvt_f32_f16<<<(D_ * D_ + 255) / 256, 256, 0, stream>>>(Wq, Wqh, D_ * D_);
  cvt_f32_f16<<<(D_ * D_ + 255) / 256, 256, 0, stream>>>(Wk, Wkh, D_ * D_);
  cvt_f32_f16<<<(D_ * D_ + 255) / 256, 256, 0, stream>>>(Wv, Wvh, D_ * D_);
  cvt_f32_f16<<<(D_ * D_ + 255) / 256, 256, 0, stream>>>(Wo, Woh, D_ * D_);
  cvt_f32_f16<<<(D_ * DH_ + 255) / 256, 256, 0, stream>>>(Wt1, Wt1h, D_ * DH_);

  // 2) edge bitmask
  zero_u32<<<(N_ * NW32 + 255) / 256, 256, 0, stream>>>(ebm, N_ * NW32);
  scatter_edges<<<(E_ + 255) / 256, 256, 0, stream>>>(ei, ebm);

  // 3) projections + topo hidden (WMMA GEMMs, TDM-staged)
  dim3 g512(M_ / 64, D_ / 64), g256(M_ / 64, DH_ / 64);
  gemm_wmma<0, 1><<<g512, 128, 0, stream>>>(Xh, Wqh, bq, Qh, nullptr, D_, D_);
  gemm_wmma<0, 1><<<g512, 128, 0, stream>>>(Xh, Wkh, bk, Kh, nullptr, D_, D_);
  gemm_wmma<0, 1><<<g512, 128, 0, stream>>>(Xh, Wvh, bv, Vh, nullptr, D_, D_);
  gemm_wmma<1, 1><<<g256, 128, 0, stream>>>(Xh, Wt1h, bt1, Hid, nullptr, D_, DH_);

  // 4) topo scores + exact top-k column mask
  topo_reduce<<<(M_ + 255) / 256, 256, 0, stream>>>(Hid, Wt2, bt2, scr);
  rank_topk<<<B_, 256, 0, stream>>>(scr, cms);

  // 5) mask_exp output (second tuple element), written directly
  float* outp = (float*)d_out;
  float* maskout = outp + (size_t)M_ * D_;
  write_mask<<<dim3(N_, B_), 256, 0, stream>>>(ebm, cms, maskout);

  // 6) fused flash attention -> O (f16), then final projection -> out (f32)
  attn_wmma<<<dim3(N_ / 64, H_, B_), 128, 0, stream>>>(Qh, Kh, Vh, ebm, cms, Oh);
  gemm_wmma<0, 0><<<g512, 128, 0, stream>>>(Oh, Woh, bo, nullptr, outp, D_, D_);
}
